// NeuralOdeWrapper_51496657879461
// MI455X (gfx1250) — matrix operand order repaired and checked by reading, not compiled, not run on hardware
//
#include <hip/hip_runtime.h>
#include <hip/hip_bf16.h>
#include <math.h>

// ---------------------------------------------------------------------------
// Heun ODE scan for Simple_MLP on gfx1250 (MI455X).
//   B=512 rows, D=64 state, P=16 params, T=500 steps, H=128 hidden.
//   One workgroup (256 threads = 8 wave32) owns 16 batch rows for the entire
//   scan: weights live in VGPR WMMA fragments, state lives in LDS (f32),
//   activations stage through LDS as f16, matmuls use v_wmma_f32_16x16x32_f16
//   (f32 accumulation). Latency-bound: dependent chain of 500*2*3 GEMM stages.
// ---------------------------------------------------------------------------

typedef __attribute__((ext_vector_type(16))) _Float16 v16h;
typedef __attribute__((ext_vector_type(8)))  float    v8f;

namespace {

constexpr int Bb  = 512;
constexpr int Dd  = 64;
constexpr int Pp  = 16;
constexpr int Tt  = 500;
constexpr int Hh  = 128;
constexpr int KIN = Dd + Pp;   // 80 (padded to 96 = 3 K-chunks of 32)
constexpr int Z0S = 104;       // staging row stride in halves (mult of 8 -> 16B aligned rows)
constexpr int Z1S = 136;

union HFrag { v16h v; _Float16 h[16]; };
union FAcc  { v8f  v; float    f[8];  };

// Branch-free tanh on the critical chain. Prefer the gfx1250 hardware
// transcendental if the toolchain exposes it; otherwise exp2+rcp identity:
//   tanh(x) = 1 - 2/(exp2(2*log2e*x) + 1)   (exact at +-1 saturation)
__device__ inline float fast_tanhf(float x) {
#if __has_builtin(__builtin_amdgcn_tanhf)
    return __builtin_amdgcn_tanhf(x);
#elif __has_builtin(__builtin_amdgcn_tanh_f32)
    return __builtin_amdgcn_tanh_f32(x);
#elif __has_builtin(__builtin_amdgcn_exp2f) && __has_builtin(__builtin_amdgcn_rcpf)
    float e = __builtin_amdgcn_exp2f(x * 2.8853900817779268f); // 2*log2(e)
    return 1.0f - 2.0f * __builtin_amdgcn_rcpf(e + 1.0f);
#else
    return tanhf(x);
#endif
}

// A-matrix fragment (16x32 f16), contiguous K order (matches B fragments):
//   m = lane%16 ; kb = (lane<16)?0:16
//   elem e in 0..15 -> K = kbase + kb + e
__device__ inline v16h load_a(const _Float16* Z, int stride, int m, int kbase, int kb) {
    HFrag a;
    const _Float16* r = Z + m * stride + kbase + kb;
#pragma unroll
    for (int e = 0; e < 8; ++e) a.h[e] = r[e];
#pragma unroll
    for (int e = 0; e < 8; ++e) a.h[8 + e] = r[8 + e];
    return a.v;
}

template <int NC>
__device__ inline v8f mlp_layer(const _Float16* Z, int stride, int m, int kb,
                                const v16h* wf) {
    v8f acc = {};
#pragma unroll
    for (int c = 0; c < NC; ++c) {
        v16h a = load_a(Z, stride, m, c * 32, kb);
        acc = __builtin_amdgcn_wmma_f32_16x16x32_f16(
            /*neg_a=*/false, a, /*neg_b=*/false, wf[c],
            /*c_mod=*/(short)0, acc, /*reuse_a=*/false, /*reuse_b=*/false);
    }
    return acc;
}

// One dfun evaluation: Z0 = [state|p] f16 staging (already built, barrier done).
// Returns D-tile fragment (+bias) for waves 0..3; zeros otherwise.
__device__ inline v8f mlp_eval(const _Float16* Z0, _Float16* Z1, _Float16* Z2,
                               int mrow, int khalf, int m_base, int ncol, int wave,
                               const v16h* w0f, const v16h* w1f, const v16h* w2f,
                               float b0v, float b1v, float b2v) {
    // layer 0: [16x80(pad96)] x [80x128]
    FAcc h0; h0.v = mlp_layer<3>(Z0, Z0S, mrow, khalf, w0f);
#pragma unroll
    for (int i = 0; i < 8; ++i)
        Z1[(m_base + i) * Z1S + ncol] = (_Float16)fast_tanhf(h0.f[i] + b0v);
    __syncthreads();
    // layer 1: [16x128] x [128x128]
    FAcc h1; h1.v = mlp_layer<4>(Z1, Z1S, mrow, khalf, w1f);
#pragma unroll
    for (int i = 0; i < 8; ++i)
        Z2[(m_base + i) * Z1S + ncol] = (_Float16)fast_tanhf(h1.f[i] + b1v);
    __syncthreads();
    // layer 2: [16x128] x [128x64] -- only 4 N-tiles, waves 0..3
    FAcc d; v8f z = {}; d.v = z;
    if (wave < 4) {
        d.v = mlp_layer<4>(Z2, Z1S, mrow, khalf, w2f);
#pragma unroll
        for (int i = 0; i < 8; ++i) d.f[i] += b2v;
    }
    return d.v;
}

} // namespace

__global__ __launch_bounds__(256)
void node_heun_kernel(const float* __restrict__ x,  const float* __restrict__ p,
                      const float* __restrict__ W0, const float* __restrict__ b0,
                      const float* __restrict__ W1, const float* __restrict__ b1,
                      const float* __restrict__ W2, const float* __restrict__ b2,
                      float* __restrict__ out) {
    __shared__ float     S[16][Dd];          // f32 carry state (16 rows x 64)
    __shared__ _Float16  Z0[16 * Z0S];       // [state | p | zero-pad] f16 staging
    __shared__ _Float16  Z1[16 * Z1S];       // hidden-1 staging
    __shared__ _Float16  Z2[16 * Z1S];       // hidden-2 staging

    const int tid  = threadIdx.x;
    const int lane = tid & 31;
    const int wave = tid >> 5;               // 0..7
    const int r0   = blockIdx.x * 16;        // base batch row

    const int ln16   = lane & 15;
    const int khalf  = (lane < 16) ? 0 : 16; // A-fragment K half offset
    const int kbB    = (lane < 16) ? 0 : 16; // B-fragment K half offset
    const int mrow   = ln16;                 // A-fragment row
    const int m_base = (lane < 16) ? 0 : 8;  // C/D M offset
    const int ncol0  = wave * 16 + ln16;     // this lane's output column in H

    // ---- Load weight fragments once; resident in VGPRs for all 500 steps ----
    v16h w0f[3], w1f[4], w2f[4];
#pragma unroll
    for (int c = 0; c < 3; ++c) {            // W0: [80 x 128], K padded to 96
        HFrag f;
#pragma unroll
        for (int e = 0; e < 16; ++e) {
            int k = c * 32 + kbB + e;
            f.h[e] = (k < KIN) ? (_Float16)W0[k * Hh + ncol0] : (_Float16)0.f;
        }
        w0f[c] = f.v;
    }
#pragma unroll
    for (int c = 0; c < 4; ++c) {            // W1: [128 x 128]
        HFrag f;
#pragma unroll
        for (int e = 0; e < 16; ++e)
            f.h[e] = (_Float16)W1[(c * 32 + kbB + e) * Hh + ncol0];
        w1f[c] = f.v;
    }
    float b2v = 0.f;
    if (wave < 4) {                          // W2: [128 x 64], 4 N-tiles
#pragma unroll
        for (int c = 0; c < 4; ++c) {
            HFrag f;
#pragma unroll
            for (int e = 0; e < 16; ++e)
                f.h[e] = (_Float16)W2[(c * 32 + kbB + e) * Dd + ncol0];
            w2f[c] = f.v;
        }
        b2v = b2[ncol0];
    }
    const float b0v = b0[ncol0];
    const float b1v = b1[ncol0];

    // ---- Init state S = x[..., 0]; zero Z0 (covers the K=80..95 pad) ----
    for (int i = tid; i < 16 * Dd; i += 256) {
        int m = i >> 6, d = i & 63;
        S[m][d] = x[((r0 + m) * Dd + d) * Tt + 0];
    }
    for (int i = tid; i < 16 * Z0S; i += 256) Z0[i] = (_Float16)0.f;
    __syncthreads();

    // p tile coordinates for this thread (16 rows x 16 params, one elem each)
    const int pm = tid >> 4, pj = tid & 15;
    const float* prow = &p[((r0 + pm) * Pp + pj) * Tt];

    // ---- Heun scan over T; emits the pre-step state (carry) at each t ----
    for (int t = 0; t < Tt; ++t) {
        // issue the p_t load first so it overlaps the out/staging stores
        float pv = prow[t];
        if (t + 1 < Tt) __builtin_prefetch(&prow[t + 1], 0, 3); // global_prefetch_b8

        // emit state and rebuild f16 staging (state columns)
        for (int i = tid; i < 16 * Dd; i += 256) {
            int m = i >> 6, d = i & 63;
            float s = S[m][d];
            out[((r0 + m) * Dd + d) * Tt + t] = s;   // strided, L2-resident
            Z0[m * Z0S + d] = (_Float16)s;
        }
        Z0[pm * Z0S + Dd + pj] = (_Float16)pv;
        __syncthreads();

        // d1 = dfun([c | p])
        FAcc d1; d1.v = mlp_eval(Z0, Z1, Z2, mrow, khalf, m_base, ncol0, wave,
                                 w0f, w1f, w2f, b0v, b1v, b2v);

        // xi = c + d1  ->  Z0 state columns (waves 0..3 own 16-wide D slices)
        if (wave < 4) {
#pragma unroll
            for (int i = 0; i < 8; ++i) {
                int m = m_base + i;
                Z0[m * Z0S + ncol0] = (_Float16)(S[m][ncol0] + d1.f[i]);
            }
        }
        __syncthreads();

        // d2 = dfun([xi | p])
        FAcc d2; d2.v = mlp_eval(Z0, Z1, Z2, mrow, khalf, m_base, ncol0, wave,
                                 w0f, w1f, w2f, b0v, b1v, b2v);

        // c <- c + 0.5*(d1 + d2)   (dt = 1.0)
        if (wave < 4) {
#pragma unroll
            for (int i = 0; i < 8; ++i) {
                int m = m_base + i;
                S[m][ncol0] += 0.5f * (d1.f[i] + d2.f[i]);
            }
        }
        __syncthreads();
    }
}

extern "C" void kernel_launch(void* const* d_in, const int* in_sizes, int n_in,
                              void* d_out, int out_size, void* d_ws, size_t ws_size,
                              hipStream_t stream) {
    (void)in_sizes; (void)n_in; (void)out_size; (void)d_ws; (void)ws_size;
    const float* x  = (const float*)d_in[0];
    const float* p  = (const float*)d_in[1];
    // d_in[2] = i_ext: scanned by the reference but never used by Simple_MLP.
    const float* W0 = (const float*)d_in[3];
    const float* b0 = (const float*)d_in[4];
    const float* W1 = (const float*)d_in[5];
    const float* b1 = (const float*)d_in[6];
    const float* W2 = (const float*)d_in[7];
    const float* b2 = (const float*)d_in[8];

    node_heun_kernel<<<dim3(Bb / 16), dim3(256), 0, stream>>>(
        x, p, W0, b0, W1, b1, W2, b2, (float*)d_out);
}
